// MLPDecoder_89215060672898
// MI455X (gfx1250) — matrix operand order repaired and checked by reading, not compiled, not run on hardware
//
#include <hip/hip_runtime.h>

typedef __attribute__((ext_vector_type(16))) __bf16 v16bf;
typedef __attribute__((ext_vector_type(8)))  __bf16 v8bf;
typedef __attribute__((ext_vector_type(8)))  float  v8f;

#define WMMA_BF16(a, b, c) \
  __builtin_amdgcn_wmma_f32_16x16x32_bf16(false, (a), false, (b), (short)0, (c), false, false)

__device__ __forceinline__ float lrelu(float x) { return x > 0.f ? x : 0.01f * x; }

// Load A-fragment (16x32 bf16) from a row-major bf16 LDS tile.
// lane<16: K = {0..7, 16..23}+kt*32 ; lane>=16: +8.  Two 16B loads.
__device__ __forceinline__ v16bf ldsA(const __bf16* base, int stride, int row, int k0) {
  v8bf lo = *(const v8bf*)(base + row * stride + k0);
  v8bf hi = *(const v8bf*)(base + row * stride + k0 + 16);
  return __builtin_shufflevector(lo, hi, 0, 1, 2, 3, 4, 5, 6, 7, 8, 9, 10, 11, 12, 13, 14, 15);
}

// Pre-swizzle a (K x N) f32 weight matrix into B-fragment order (bf16):
// frag(kt,nt): lane l holds K values k = kt*32 + 8*(l>=16) + (e&7) + 16*(e>=8) for n = nt*16 + l%16.
__global__ __launch_bounds__(256) void swizzle_kernel(const float* __restrict__ W,
                                                      __bf16* __restrict__ dst, int K, int N) {
  int idx = blockIdx.x * 256 + threadIdx.x;
  if (idx >= K * N) return;
  int e = idx & 15;
  int lane = (idx >> 4) & 31;
  int frag = idx >> 9;
  int NT = N >> 4;
  int nt = frag % NT;
  int kt = frag / NT;
  int n = nt * 16 + (lane & 15);
  int k = kt * 32 + (e & 7) + ((lane >> 4) << 3) + ((e >> 3) << 4);
  dst[idx] = (__bf16)W[k * N + n];
}

// One block per (b, t0..t0+1, receiver a): 2 time steps x (31 edges + 1 gated pad) = 64 rows.
// 8 waves: wave w owns N-tiles {2w, 2w+1} and ALL 4 M-tiles -> each B frag loaded once, used 4x.
__global__ __launch_bounds__(256) void msg_kernel(
    const float* __restrict__ inputs, const float* __restrict__ state,
    const float* __restrict__ rel_type,
    const __bf16* __restrict__ w1f, const float* __restrict__ b1,
    const __bf16* __restrict__ w2f, const float* __restrict__ b2,
    float* __restrict__ agg) {
  const int a = blockIdx.x, b = blockIdx.z;
  const int t0 = blockIdx.y * 2;
  const int tid = threadIdx.x;
  const int lane = tid & 31, wv = tid >> 5;
  const int hi = lane >> 4, ln = lane & 15;
  const int nt0 = wv * 2;

  __shared__ alignas(32) __bf16 pm[64 * 128];   // pre_msg rows: [tl*32 + edge][128]
  __shared__ alignas(32) __bf16 h1[64 * 256];   // hidden
  __shared__ float gates[3 * 64];
  __shared__ float red[8 * 256];                // row-reduction accumulator

  // Stage pre_msg = [send | recv] as bf16. Pad rows (edge 31) duplicate edge 30 (gate=0).
#pragma unroll
  for (int ss = 0; ss < 2; ++ss) {
    int slot = tid + ss * 256;       // 512 slots = 64 rows x 8 segs
    int row = slot >> 3, seg = slot & 7;
    int tl = row >> 5, r32 = row & 31;
    int j = r32 < 31 ? r32 : 30;
    int snd = j + (j >= a);
    int tt = t0 + tl;
    const float* src = (seg < 4)
        ? inputs + (((b * 32 + snd) * 64 + tt) * 64 + seg * 16)
        : inputs + (((b * 32 + a) * 64 + tt) * 64 + (seg - 4) * 16);
#pragma unroll
    for (int c = 0; c < 16; ++c) pm[row * 128 + seg * 16 + c] = (__bf16)src[c];
  }
  // Gates g[i][tl*32+row] = sum_s rel_type[b,r,s,i+1] * state[b,send,t,s]
  if (tid < 192) {
    int i = tid / 64, row = tid & 63;
    int tl = row >> 5, r32 = row & 31;
    float g = 0.f;
    if (r32 < 31) {
      int r = a * 31 + r32;
      int snd = r32 + (r32 >= a);
#pragma unroll
      for (int s = 0; s < 4; ++s)
        g += rel_type[((b * 992 + r) * 4 + s) * 4 + (i + 1)] *
             state[((b * 32 + snd) * 64 + (t0 + tl)) * 4 + s];
    }
    gates[i * 64 + row] = g;
  }
  // zero the reduction accumulator
#pragma unroll
  for (int q = 0; q < 8; ++q) red[q * 256 + tid] = 0.f;
  __syncthreads();

  for (int i = 0; i < 3; ++i) {  // edge types 1..3
    const __bf16* W1 = w1f + i * (128 * 256);
    const __bf16* W2 = w2f + i * (256 * 256);
    // fc1: 64x128 @ 128x256
    v8f acc[4][2];
#pragma unroll
    for (int mt = 0; mt < 4; ++mt)
#pragma unroll
      for (int tp = 0; tp < 2; ++tp)
#pragma unroll
        for (int j = 0; j < 8; ++j) acc[mt][tp][j] = 0.f;
#pragma unroll
    for (int kt = 0; kt < 4; ++kt) {
      v16bf bf0 = *(const v16bf*)(W1 + (kt * 16 + nt0) * 512 + lane * 16);
      v16bf bf1 = *(const v16bf*)(W1 + (kt * 16 + nt0 + 1) * 512 + lane * 16);
#pragma unroll
      for (int mt = 0; mt < 4; ++mt) {
        v16bf af = ldsA(pm, 128, mt * 16 + ln, kt * 32 + hi * 8);
        acc[mt][0] = WMMA_BF16(af, bf0, acc[mt][0]);
        acc[mt][1] = WMMA_BF16(af, bf1, acc[mt][1]);
      }
    }
    {
      float bn0 = b1[(i + 1) * 256 + nt0 * 16 + ln];
      float bn1 = b1[(i + 1) * 256 + (nt0 + 1) * 16 + ln];
#pragma unroll
      for (int mt = 0; mt < 4; ++mt)
#pragma unroll
        for (int j = 0; j < 8; ++j) {
          int rr = mt * 16 + j + hi * 8;
          h1[rr * 256 + nt0 * 16 + ln] = (__bf16)lrelu(acc[mt][0][j] + bn0);
          h1[rr * 256 + (nt0 + 1) * 16 + ln] = (__bf16)lrelu(acc[mt][1][j] + bn1);
        }
    }
    __syncthreads();
    // fc2: 64x256 @ 256x256, then gate + row-reduce into red
    v8f a2[4][2];
#pragma unroll
    for (int mt = 0; mt < 4; ++mt)
#pragma unroll
      for (int tp = 0; tp < 2; ++tp)
#pragma unroll
        for (int j = 0; j < 8; ++j) a2[mt][tp][j] = 0.f;
#pragma unroll
    for (int kt = 0; kt < 8; ++kt) {
      v16bf bf0 = *(const v16bf*)(W2 + (kt * 16 + nt0) * 512 + lane * 16);
      v16bf bf1 = *(const v16bf*)(W2 + (kt * 16 + nt0 + 1) * 512 + lane * 16);
#pragma unroll
      for (int mt = 0; mt < 4; ++mt) {
        v16bf af = ldsA(h1, 256, mt * 16 + ln, kt * 32 + hi * 8);
        a2[mt][0] = WMMA_BF16(af, bf0, a2[mt][0]);
        a2[mt][1] = WMMA_BF16(af, bf1, a2[mt][1]);
      }
    }
    {
      float c0 = b2[(i + 1) * 256 + nt0 * 16 + ln];
      float c1 = b2[(i + 1) * 256 + (nt0 + 1) * 16 + ln];
#pragma unroll
      for (int mt = 0; mt < 4; ++mt) {
        float p0 = 0.f, p1 = 0.f;
#pragma unroll
        for (int j = 0; j < 8; ++j) {
          float g = gates[i * 64 + mt * 16 + j + hi * 8];
          p0 += lrelu(a2[mt][0][j] + c0) * g;
          p1 += lrelu(a2[mt][1][j] + c1) * g;
        }
        red[(mt * 2 + hi) * 256 + nt0 * 16 + ln] += p0;
        red[(mt * 2 + hi) * 256 + (nt0 + 1) * 16 + ln] += p1;
      }
    }
    __syncthreads();  // before next type rewrites h1 (and protects red RMW ownership epochs)
  }

  // Final: agg[b, t0+tl, a, :] = sum over the 4 red slices of that tl
#pragma unroll
  for (int tl = 0; tl < 2; ++tl) {
    float v = red[(tl * 4 + 0) * 256 + tid] + red[(tl * 4 + 1) * 256 + tid] +
              red[(tl * 4 + 2) * 256 + tid] + red[(tl * 4 + 3) * 256 + tid];
    agg[((b * 64 + (t0 + tl)) * 32 + a) * 256 + tid] = v;
  }
}

// Node MLP: 32 rows of (b,t,a) per block; aug=[x|agg] (320) -> 256 -> 256 -> 64 (+x residual).
// Wave w owns N-tiles {2w, 2w+1} and both M-tiles (B frags loaded once per block).
__global__ __launch_bounds__(256) void node_kernel(
    const float* __restrict__ inputs, const float* __restrict__ agg,
    const __bf16* __restrict__ w1f, const float* __restrict__ b1,
    const __bf16* __restrict__ w2f, const float* __restrict__ b2,
    const __bf16* __restrict__ w3f, const float* __restrict__ b3,
    float* __restrict__ out) {
  const int tid = threadIdx.x;
  const int lane = tid & 31, wv = tid >> 5;
  const int hi = lane >> 4, ln = lane & 15;
  const int nt0 = wv * 2;
  const int row0 = blockIdx.x * 32;  // row index = (b*64 + t)*32 + a

  __shared__ alignas(32) __bf16 aug[32 * 320];
  __shared__ alignas(32) __bf16 hA[32 * 256];
  __shared__ alignas(32) __bf16 hB[32 * 256];

  {
    int row = tid >> 3, seg = tid & 7;
    int gr = row0 + row;
    int aa = gr & 31, t = (gr >> 5) & 63, bb = gr >> 11;
    const float* sx = inputs + (((bb * 32 + aa) * 64 + t) * 64 + seg * 8);
#pragma unroll
    for (int c = 0; c < 8; ++c) aug[row * 320 + seg * 8 + c] = (__bf16)sx[c];
    const float* sg = agg + gr * 256 + seg * 32;
#pragma unroll
    for (int c = 0; c < 32; ++c) aug[row * 320 + 64 + seg * 32 + c] = (__bf16)sg[c];
  }
  __syncthreads();

  // fc1: K=320
  {
    v8f acc[2][2];
#pragma unroll
    for (int mt = 0; mt < 2; ++mt)
#pragma unroll
      for (int tp = 0; tp < 2; ++tp)
#pragma unroll
        for (int j = 0; j < 8; ++j) acc[mt][tp][j] = 0.f;
#pragma unroll
    for (int kt = 0; kt < 10; ++kt) {
      v16bf bf0 = *(const v16bf*)(w1f + (kt * 16 + nt0) * 512 + lane * 16);
      v16bf bf1 = *(const v16bf*)(w1f + (kt * 16 + nt0 + 1) * 512 + lane * 16);
#pragma unroll
      for (int mt = 0; mt < 2; ++mt) {
        v16bf af = ldsA(aug, 320, mt * 16 + ln, kt * 32 + hi * 8);
        acc[mt][0] = WMMA_BF16(af, bf0, acc[mt][0]);
        acc[mt][1] = WMMA_BF16(af, bf1, acc[mt][1]);
      }
    }
    float bn0 = b1[nt0 * 16 + ln], bn1 = b1[(nt0 + 1) * 16 + ln];
#pragma unroll
    for (int mt = 0; mt < 2; ++mt)
#pragma unroll
      for (int j = 0; j < 8; ++j) {
        int rr = mt * 16 + j + hi * 8;
        hA[rr * 256 + nt0 * 16 + ln] = (__bf16)lrelu(acc[mt][0][j] + bn0);
        hA[rr * 256 + (nt0 + 1) * 16 + ln] = (__bf16)lrelu(acc[mt][1][j] + bn1);
      }
  }
  __syncthreads();
  // fc2: K=256
  {
    v8f acc[2][2];
#pragma unroll
    for (int mt = 0; mt < 2; ++mt)
#pragma unroll
      for (int tp = 0; tp < 2; ++tp)
#pragma unroll
        for (int j = 0; j < 8; ++j) acc[mt][tp][j] = 0.f;
#pragma unroll
    for (int kt = 0; kt < 8; ++kt) {
      v16bf bf0 = *(const v16bf*)(w2f + (kt * 16 + nt0) * 512 + lane * 16);
      v16bf bf1 = *(const v16bf*)(w2f + (kt * 16 + nt0 + 1) * 512 + lane * 16);
#pragma unroll
      for (int mt = 0; mt < 2; ++mt) {
        v16bf af = ldsA(hA, 256, mt * 16 + ln, kt * 32 + hi * 8);
        acc[mt][0] = WMMA_BF16(af, bf0, acc[mt][0]);
        acc[mt][1] = WMMA_BF16(af, bf1, acc[mt][1]);
      }
    }
    float bn0 = b2[nt0 * 16 + ln], bn1 = b2[(nt0 + 1) * 16 + ln];
#pragma unroll
    for (int mt = 0; mt < 2; ++mt)
#pragma unroll
      for (int j = 0; j < 8; ++j) {
        int rr = mt * 16 + j + hi * 8;
        hB[rr * 256 + nt0 * 16 + ln] = (__bf16)lrelu(acc[mt][0][j] + bn0);
        hB[rr * 256 + (nt0 + 1) * 16 + ln] = (__bf16)lrelu(acc[mt][1][j] + bn1);
      }
  }
  __syncthreads();
  // fc3: K=256, N=64 (NT=4) -> one 16x16 tile per wave; residual + slice + transpose in store.
  {
    int mt3 = wv >> 2, nt3 = wv & 3;
    int m0 = mt3 * 16;
    v8f a3;
#pragma unroll
    for (int j = 0; j < 8; ++j) a3[j] = 0.f;
#pragma unroll
    for (int kt = 0; kt < 8; ++kt) {
      v16bf af = ldsA(hB, 256, m0 + ln, kt * 32 + hi * 8);
      v16bf bf = *(const v16bf*)(w3f + (kt * 4 + nt3) * 512 + lane * 16);
      a3 = WMMA_BF16(af, bf, a3);
    }
    float bn = b3[nt3 * 16 + ln];
    int d = nt3 * 16 + ln;
#pragma unroll
    for (int j = 0; j < 8; ++j) {
      int gr = row0 + m0 + j + hi * 8;
      int aa = gr & 31, t = (gr >> 5) & 63, bb = gr >> 11;
      if (t < 63) {
        float xv = inputs[((bb * 32 + aa) * 64 + t) * 64 + d];
        out[((bb * 32 + aa) * 63 + t) * 64 + d] = xv + a3[j] + bn;
      }
    }
  }
}

extern "C" void kernel_launch(void* const* d_in, const int* in_sizes, int n_in,
                              void* d_out, int out_size, void* d_ws, size_t ws_size,
                              hipStream_t stream) {
  const float* inputs    = (const float*)d_in[0];
  const float* state     = (const float*)d_in[1];
  const float* rel_type  = (const float*)d_in[2];
  const float* msg_fc1_w = (const float*)d_in[5];
  const float* msg_fc1_b = (const float*)d_in[6];
  const float* msg_fc2_w = (const float*)d_in[7];
  const float* msg_fc2_b = (const float*)d_in[8];
  const float* out_fc1_w = (const float*)d_in[9];
  const float* out_fc1_b = (const float*)d_in[10];
  const float* out_fc2_w = (const float*)d_in[11];
  const float* out_fc2_b = (const float*)d_in[12];
  const float* out_fc3_w = (const float*)d_in[13];
  const float* out_fc3_b = (const float*)d_in[14];

  char* ws = (char*)d_ws;
  __bf16* w1f = (__bf16*)(ws + 0);        // 3*128*256 bf16 = 196608 B
  __bf16* w2f = (__bf16*)(ws + 196608);   // 3*256*256 bf16 = 393216 B
  __bf16* o1f = (__bf16*)(ws + 589824);   // 320*256 bf16   = 163840 B
  __bf16* o2f = (__bf16*)(ws + 753664);   // 256*256 bf16   = 131072 B
  __bf16* o3f = (__bf16*)(ws + 884736);   // 256*64  bf16   =  32768 B
  float*  agg = (float*)(ws + 917504);    // 8192*256 f32   = 8388608 B

  // Pre-swizzle weights into WMMA B-fragment order (bf16).
  for (int i = 1; i < 4; ++i) {
    swizzle_kernel<<<128, 256, 0, stream>>>(msg_fc1_w + i * 128 * 256,
                                            w1f + (i - 1) * 128 * 256, 128, 256);
    swizzle_kernel<<<256, 256, 0, stream>>>(msg_fc2_w + i * 256 * 256,
                                            w2f + (i - 1) * 256 * 256, 256, 256);
  }
  swizzle_kernel<<<320, 256, 0, stream>>>(out_fc1_w, o1f, 320, 256);
  swizzle_kernel<<<256, 256, 0, stream>>>(out_fc2_w, o2f, 256, 256);
  swizzle_kernel<<<64, 256, 0, stream>>>(out_fc3_w, o3f, 256, 64);

  dim3 g1(32, 32, 4);  // (a, t-pair, b)
  msg_kernel<<<g1, 256, 0, stream>>>(inputs, state, rel_type, w1f, msg_fc1_b, w2f,
                                     msg_fc2_b, agg);
  node_kernel<<<256, 256, 0, stream>>>(inputs, agg, o1f, out_fc1_b, o2f, out_fc2_b, o3f,
                                       out_fc3_b, (float*)d_out);
}
